// ATT_MIL_29051158790227
// MI455X (gfx1250) — matrix operand order repaired and checked by reading, not compiled, not run on hardware
//
#include <hip/hip_runtime.h>
#include <hip/hip_bf16.h>
#include <math.h>

typedef __attribute__((ext_vector_type(16))) _Float16 v16h;
typedef __attribute__((ext_vector_type(8)))  _Float16 v8h;
typedef __attribute__((ext_vector_type(8)))  float    v8f;
typedef __attribute__((ext_vector_type(4)))  float    v4f;

// ---------------------------------------------------------------------------
// helpers
// ---------------------------------------------------------------------------
__device__ __forceinline__ v8f wmma_f16(v16h a, v16h b, v8f c) {
    // D = A(16x32 f16) * B(32x16 f16) + C(16x16 f32)
    return __builtin_amdgcn_wmma_f32_16x16x32_f16(
        false, a, false, b, (short)0, c, false, false);
}

// load 8 consecutive f32 from p, convert to f16, place at d[o..o+7]
__device__ __forceinline__ void load8h(const float* __restrict__ p, v16h& d, int o) {
    v4f x0 = *reinterpret_cast<const v4f*>(p);
    v4f x1 = *reinterpret_cast<const v4f*>(p + 4);
    d[o+0]=(_Float16)x0[0]; d[o+1]=(_Float16)x0[1];
    d[o+2]=(_Float16)x0[2]; d[o+3]=(_Float16)x0[3];
    d[o+4]=(_Float16)x1[0]; d[o+5]=(_Float16)x1[1];
    d[o+6]=(_Float16)x1[2]; d[o+7]=(_Float16)x1[3];
}

// load 8 consecutive f16 from p (16B aligned), place at d[o..o+7]
__device__ __forceinline__ void load8hh(const _Float16* __restrict__ p, v16h& d, int o) {
    v8h x = *reinterpret_cast<const v8h*>(p);
    d[o+0]=x[0]; d[o+1]=x[1]; d[o+2]=x[2]; d[o+3]=x[3];
    d[o+4]=x[4]; d[o+5]=x[5]; d[o+6]=x[6]; d[o+7]=x[7];
}

// f32 -> f16 conversion (weights shadow copies)
__global__ __launch_bounds__(256)
void f32_to_f16(const float* __restrict__ src, _Float16* __restrict__ dst, int n)
{
    int i = blockIdx.x * 256 + threadIdx.x;
    if (i < n) dst[i] = (_Float16)src[i];
}

// ---------------------------------------------------------------------------
// WMMA GEMM:  C[Mr,Nc] = A[Mr,K](f32) * W16[Nc,K](f16)^T (+bias)(+res)(relu)
// Optional f32 output C and/or f16 output C16.
// One wave computes a 16x64 tile. Mr%16==0, Nc%64==0, K%32==0.
// ---------------------------------------------------------------------------
__global__ __launch_bounds__(256)
void gemm_wmma(const float* __restrict__ A, const _Float16* __restrict__ W16,
               const float* __restrict__ bias, const float* __restrict__ res,
               float* __restrict__ C, _Float16* __restrict__ C16,
               int Mr, int Nc, int K, int relu)
{
    int wave = (int)((blockIdx.x * blockDim.x + threadIdx.x) >> 5);
    int lane = threadIdx.x & 31;
    int ctiles = Nc >> 6;
    int total  = (Mr >> 4) * ctiles;
    if (wave >= total) return;                 // whole-wave uniform exit
    int rt = wave / ctiles, ct = wave % ctiles;
    int lr = lane & 15, g = lane >> 4;

    v8f z = {};
    v8f acc[4]; acc[0]=z; acc[1]=z; acc[2]=z; acc[3]=z;

    const float* arow = A + (size_t)(rt*16 + lr) * K;
    for (int kb = 0; kb < K; kb += 32) {
        v16h af;
        load8h(arow + kb + 16*g,     af, 0);   // K = kb + 16g .. +7
        load8h(arow + kb + 16*g + 8, af, 8);   // K = kb + 16g + 8 .. +15
#pragma unroll
        for (int j = 0; j < 4; ++j) {
            int wr = ct*64 + j*16 + lr;        // output column == weight row
            // 16 consecutive f16 (32B aligned): K_local = 16g + 0..15
            v16h bf = *reinterpret_cast<const v16h*>(
                W16 + (size_t)wr * K + kb + 16*g);
            acc[j] = wmma_f16(af, bf, acc[j]);
        }
    }
#pragma unroll
    for (int j = 0; j < 4; ++j) {
        int col = ct*64 + j*16 + lr;
        float bv = bias ? bias[col] : 0.f;
#pragma unroll
        for (int r = 0; r < 8; ++r) {
            int orow = rt*16 + r + 8*g;
            float v = acc[j][r] + bv;
            if (res)  v += res[(size_t)orow * Nc + col];
            if (relu) v = fmaxf(v, 0.f);
            size_t idx = (size_t)orow * Nc + col;
            if (C)   C[idx]   = v;
            if (C16) C16[idx] = (_Float16)v;
        }
    }
}

// ---------------------------------------------------------------------------
// Flash attention over f16 qkv[4096,1536] (q|k|v each 512 = 8 heads x 64)
// one wave (block=32) per (head, 16-query block); online softmax, 4096 keys.
// 1/sqrt(64) applied to raw scores (post-WMMA) instead of Q.
// ---------------------------------------------------------------------------
__global__ __launch_bounds__(32)
void attn_flash(const _Float16* __restrict__ qkv, float* __restrict__ AO)
{
    __shared__ float pbuf[16 * 32];
    const int SQ = 1536;
    int w    = blockIdx.x;
    int head = w >> 8;            // 4096/16 = 256 q-blocks per head
    int qt   = w & 255;
    int lane = threadIdx.x, lr = lane & 15, g = lane >> 4;
    int q0   = qt * 16;

    // Q fragments, head dims 0..31 and 32..63 (A-layout: two 8-half runs)
    const _Float16* qrow = qkv + (size_t)(q0 + lr) * SQ + head * 64;
    v16h a0, a1;
    load8hh(qrow +      16*g, a0, 0);
    load8hh(qrow +  8 + 16*g, a0, 8);
    load8hh(qrow + 32 + 16*g, a1, 0);
    load8hh(qrow + 40 + 16*g, a1, 8);

    v8f z = {};
    v8f o[4]; o[0]=z; o[1]=z; o[2]=z; o[3]=z;
    float rm[8], rs[8];
#pragma unroll
    for (int r = 0; r < 8; ++r) { rm[r] = -3.0e38f; rs[r] = 0.f; }

    for (int kb = 0; kb < 4096; kb += 32) {
        v8f s[2];
#pragma unroll
        for (int t = 0; t < 2; ++t) {
            int key = kb + 16*t + lr;
            const _Float16* krow = qkv + (size_t)key * SQ + 512 + head * 64 + 16*g;
            v16h b0 = *reinterpret_cast<const v16h*>(krow);       // dims 16g..+15
            v16h b1 = *reinterpret_cast<const v16h*>(krow + 32);  // dims 32+16g..+15
            v8f c = z;
            c = wmma_f16(a0, b0, c);
            c = wmma_f16(a1, b1, c);
            s[t] = c;
        }
        // online softmax per row (rows r+8g live in 16-lane groups)
#pragma unroll
        for (int r = 0; r < 8; ++r) {
            float v0 = s[0][r] * 0.125f, v1 = s[1][r] * 0.125f;
            float mx = fmaxf(v0, v1);
            mx = fmaxf(mx, __shfl_xor(mx, 1, 32));
            mx = fmaxf(mx, __shfl_xor(mx, 2, 32));
            mx = fmaxf(mx, __shfl_xor(mx, 4, 32));
            mx = fmaxf(mx, __shfl_xor(mx, 8, 32));
            float mnew = fmaxf(rm[r], mx);
            float sc   = __expf(rm[r] - mnew);
            float p0   = __expf(v0 - mnew);
            float p1   = __expf(v1 - mnew);
            float ps   = p0 + p1;
            ps += __shfl_xor(ps, 1, 32);
            ps += __shfl_xor(ps, 2, 32);
            ps += __shfl_xor(ps, 4, 32);
            ps += __shfl_xor(ps, 8, 32);
            rs[r] = rs[r] * sc + ps;
            rm[r] = mnew;
            o[0][r] *= sc; o[1][r] *= sc; o[2][r] *= sc; o[3][r] *= sc;
            pbuf[(r + 8*g) * 32 +      lr] = p0;
            pbuf[(r + 8*g) * 32 + 16 + lr] = p1;
        }
        __syncthreads();                          // single-wave WG: cheap fence
        // rebuild P as an A-fragment (row = lane%16, K = 32 keys)
        const float* prow = &pbuf[lr * 32];
        v16h pa;
        load8h(prow +     16*g, pa, 0);
        load8h(prow + 8 + 16*g, pa, 8);
        __syncthreads();
        // O += P * V   (V tile 32x64, 4 N-tiles)
#pragma unroll
        for (int j = 0; j < 4; ++j) {
            const _Float16* vbase = qkv + 1024 + head * 64 + j*16 + lr;
            v16h bv;
#pragma unroll
            for (int kk = 0; kk < 16; ++kk)
                bv[kk] = vbase[(size_t)(kb + 16*g + kk) * SQ];
            o[j] = wmma_f16(pa, bv, o[j]);
        }
    }
    // write O / row-sum
#pragma unroll
    for (int j = 0; j < 4; ++j)
#pragma unroll
        for (int r = 0; r < 8; ++r) {
            int row = q0 + r + 8*g;
            AO[(size_t)row * 512 + head * 64 + j*16 + lr] = o[j][r] / rs[r];
        }
}

// ---------------------------------------------------------------------------
// LayerNorm over M=512, one block per row
// ---------------------------------------------------------------------------
__global__ __launch_bounds__(256)
void lnorm(const float* __restrict__ Y, const float* __restrict__ g,
           const float* __restrict__ b, float* __restrict__ X)
{
    __shared__ float red[256];
    int row = blockIdx.x, t = threadIdx.x;
    const float* y = Y + (size_t)row * 512;
    float v0 = y[t], v1 = y[t + 256];
    red[t] = v0 + v1; __syncthreads();
    for (int s = 128; s > 0; s >>= 1) { if (t < s) red[t] += red[t+s]; __syncthreads(); }
    float mu = red[0] * (1.f/512.f); __syncthreads();
    float d0 = v0 - mu, d1 = v1 - mu;
    red[t] = d0*d0 + d1*d1; __syncthreads();
    for (int s = 128; s > 0; s >>= 1) { if (t < s) red[t] += red[t+s]; __syncthreads(); }
    float rstd = rsqrtf(red[0] * (1.f/512.f) + 1e-5f);
    X[(size_t)row*512 + t]       = d0 * rstd * g[t]       + b[t];
    X[(size_t)row*512 + t + 256] = d1 * rstd * g[t + 256] + b[t + 256];
}

// ---------------------------------------------------------------------------
// gated logits (level 1): w1[n] = sum_j tanh(tv[n,j])*sigmoid(tu[n,j])*w[j], L=256
// ---------------------------------------------------------------------------
__global__ __launch_bounds__(256)
void gated1(const float* __restrict__ tv, const float* __restrict__ tu,
            const float* __restrict__ w, float* __restrict__ w1)
{
    __shared__ float red[256];
    int row = blockIdx.x, t = threadIdx.x;
    float a  = tanhf(tv[(size_t)row*256 + t]);
    float sg = 1.f / (1.f + __expf(-tu[(size_t)row*256 + t]));
    red[t] = a * sg * w[t]; __syncthreads();
    for (int s = 128; s > 0; s >>= 1) { if (t < s) red[t] += red[t+s]; __syncthreads(); }
    if (t == 0) w1[row] = red[0];
}

// ---------------------------------------------------------------------------
// inner segment softmax (128 segments over 4096) + super_ids = segmin(ids)
// ---------------------------------------------------------------------------
__global__ __launch_bounds__(256)
void seg_soft_inner(const float* __restrict__ w1, const int* __restrict__ seg,
                    const int* __restrict__ ids, float* __restrict__ a1,
                    int* __restrict__ sup)
{
    __shared__ float red[256];
    __shared__ int   redi[256];
    __shared__ float s_m, s_s;
    int b = blockIdx.x, t = threadIdx.x;
    float mx = -3.0e38f; int mn = 0x7fffffff;
    for (int n = t; n < 4096; n += 256)
        if (seg[n] == b) { mx = fmaxf(mx, w1[n]); int v = ids[n]; mn = (v < mn) ? v : mn; }
    red[t] = mx; redi[t] = mn; __syncthreads();
    for (int s = 128; s > 0; s >>= 1) {
        if (t < s) { red[t] = fmaxf(red[t], red[t+s]);
                     redi[t] = (redi[t+s] < redi[t]) ? redi[t+s] : redi[t]; }
        __syncthreads();
    }
    if (t == 0) { s_m = red[0]; sup[b] = redi[0]; }
    __syncthreads();
    float m = s_m, sum = 0.f;
    for (int n = t; n < 4096; n += 256) if (seg[n] == b) sum += __expf(w1[n] - m);
    red[t] = sum; __syncthreads();
    for (int s = 128; s > 0; s >>= 1) { if (t < s) red[t] += red[t+s]; __syncthreads(); }
    if (t == 0) s_s = red[0];
    __syncthreads();
    float inv = 1.f / s_s;
    for (int n = t; n < 4096; n += 256)
        if (seg[n] == b) a1[n] = __expf(w1[n] - m) * inv;
}

// inner pooling: outp[b,:] = sum_{seg[n]==b} a1[n]*inst[n,:]
__global__ __launch_bounds__(256)
void pool_inner(const float* __restrict__ inst, const float* __restrict__ a1,
                const int* __restrict__ seg, float* __restrict__ outp)
{
    int b = blockIdx.x, t = threadIdx.x;
    float acc0 = 0.f, acc1 = 0.f;
    for (int n = 0; n < 4096; ++n) {
        if (seg[n] == b) {
            float a = a1[n];
            acc0 += a * inst[(size_t)n*512 + t];
            acc1 += a * inst[(size_t)n*512 + t + 256];
        }
    }
    outp[(size_t)b*512 + t]       = acc0;
    outp[(size_t)b*512 + t + 256] = acc1;
}

// gated logits (level 2) on outp[128,512]
__global__ __launch_bounds__(256)
void gated2(const float* __restrict__ outp, const float* __restrict__ V,
            const float* __restrict__ U, const float* __restrict__ w,
            float* __restrict__ w2)
{
    __shared__ float red[256];
    int i = blockIdx.x, j = threadIdx.x;
    const float* x  = outp + (size_t)i*512;
    const float* vr = V + (size_t)j*512;
    const float* ur = U + (size_t)j*512;
    float dv = 0.f, du = 0.f;
    for (int k = 0; k < 512; ++k) { dv += x[k]*vr[k]; du += x[k]*ur[k]; }
    red[j] = tanhf(dv) * (1.f / (1.f + __expf(-du))) * w[j];
    __syncthreads();
    for (int s = 128; s > 0; s >>= 1) { if (j < s) red[j] += red[j+s]; __syncthreads(); }
    if (j == 0) w2[i] = red[0];
}

// outer segment softmax: 128 inner bags -> 16 outer bags, single block
__global__ __launch_bounds__(128)
void seg_soft_outer(const float* __restrict__ w2, const int* __restrict__ sup,
                    float* __restrict__ a2)
{
    __shared__ float sw[128]; __shared__ int ss[128];
    __shared__ float sm[16], sss[16];
    int t = threadIdx.x;
    sw[t] = w2[t]; ss[t] = sup[t]; __syncthreads();
    if (t < 16) {
        float m = -3.0e38f;
        for (int i = 0; i < 128; ++i) if (ss[i] == t) m = fmaxf(m, sw[i]);
        sm[t] = m;
    }
    __syncthreads();
    if (t < 16) {
        float s = 0.f;
        for (int i = 0; i < 128; ++i) if (ss[i] == t) s += __expf(sw[i] - sm[t]);
        sss[t] = s;
    }
    __syncthreads();
    int o = ss[t];
    a2[t] = __expf(sw[t] - sm[o]) / sss[o];
}

// outer pooling: out2[o,:] = sum_{sup[i]==o} a2[i]*outp[i,:]
__global__ __launch_bounds__(256)
void pool_outer(const float* __restrict__ outp, const float* __restrict__ a2,
                const int* __restrict__ sup, float* __restrict__ out2)
{
    int o = blockIdx.x, t = threadIdx.x;
    float acc0 = 0.f, acc1 = 0.f;
    for (int i = 0; i < 128; ++i) {
        if (sup[i] == o) {
            float a = a2[i];
            acc0 += a * outp[(size_t)i*512 + t];
            acc1 += a * outp[(size_t)i*512 + t + 256];
        }
    }
    out2[(size_t)o*512 + t]       = acc0;
    out2[(size_t)o*512 + t + 256] = acc1;
}

// conv1d(1->16,k3,SAME) + relu + mean + linear + sigmoid
__global__ __launch_bounds__(256)
void cnn_head(const float* __restrict__ out2, const float* __restrict__ cw,
              const float* __restrict__ cb, const float* __restrict__ fw,
              const float* __restrict__ fb, float* __restrict__ yprob,
              float* __restrict__ yhat)
{
    __shared__ float red[256];
    __shared__ float hc[16];
    int o = blockIdx.x, t = threadIdx.x;
    const float* x = out2 + (size_t)o * 512;
    for (int c = 0; c < 16; ++c) {
        float w0 = cw[c*3+0], w1 = cw[c*3+1], w2 = cw[c*3+2], bb = cb[c];
        float part = 0.f;
        for (int p = t; p < 512; p += 256) {
            float xm = (p > 0)   ? x[p-1] : 0.f;
            float xc = x[p];
            float xp = (p < 511) ? x[p+1] : 0.f;
            part += fmaxf(bb + w0*xm + w1*xc + w2*xp, 0.f);
        }
        red[t] = part; __syncthreads();
        for (int s = 128; s > 0; s >>= 1) { if (t < s) red[t] += red[t+s]; __syncthreads(); }
        if (t == 0) hc[c] = red[0] * (1.f/512.f);
        __syncthreads();
    }
    if (t == 0) {
        float zv = fb[0];
        for (int c = 0; c < 16; ++c) zv += hc[c] * fw[c];
        float p = 1.f / (1.f + __expf(-zv));
        yprob[o] = p;
        yhat[o]  = (p >= 0.5f) ? 1.f : 0.f;
    }
}

// ---------------------------------------------------------------------------
// host side
// ---------------------------------------------------------------------------
static inline int gemm_blocks(int Mr, int Nc) {
    int waves = (Mr / 16) * (Nc / 64);
    return (waves + 7) / 8;   // 8 waves / 256-thread block
}

extern "C" void kernel_launch(void* const* d_in, const int* in_sizes, int n_in,
                              void* d_out, int out_size, void* d_ws, size_t ws_size,
                              hipStream_t stream) {
    const int N = 4096, M = 512, FF = 2048, Lg = 256;
    const float* datas   = (const float*)d_in[0];
    const int*   Seq_ids = (const int*)d_in[1];
    const int*   ids     = (const int*)d_in[2];
    const float* Wqkv    = (const float*)d_in[5];
    const float* bqkv    = (const float*)d_in[6];
    const float* Wo      = (const float*)d_in[7];
    const float* bo      = (const float*)d_in[8];
    const float* ln1g    = (const float*)d_in[9];
    const float* ln1b    = (const float*)d_in[10];
    const float* W1      = (const float*)d_in[11];
    const float* b1      = (const float*)d_in[12];
    const float* W2      = (const float*)d_in[13];
    const float* b2      = (const float*)d_in[14];
    const float* ln2g    = (const float*)d_in[15];
    const float* ln2b    = (const float*)d_in[16];
    const float* ga1V    = (const float*)d_in[17];
    const float* ga1U    = (const float*)d_in[18];
    const float* ga1w    = (const float*)d_in[19];
    const float* ga2V    = (const float*)d_in[20];
    const float* ga2U    = (const float*)d_in[21];
    const float* ga2w    = (const float*)d_in[22];
    const float* conv_w  = (const float*)d_in[23];
    const float* conv_b  = (const float*)d_in[24];
    const float* fc_w    = (const float*)d_in[25];
    const float* fc_b    = (const float*)d_in[26];

    // ---- f32 workspace ----
    float* ws   = (float*)d_ws;
    float* X    = ws;                     // 4096*512
    float* Y    = X   + (size_t)N*M;      // 4096*512
    float* AO   = Y   + (size_t)N*M;      // 4096*512
    float* T1   = AO  + (size_t)N*M;      // 4096*2048  (also tv/tu)
    float* SS   = T1  + (size_t)N*FF;
    float* w1   = SS;                     // 4096
    float* OUTP = SS + N;                 // 128*512
    float* w2v  = OUTP + 128*512;         // 128
    int*   sup  = (int*)(w2v + 128);      // 128
    // ---- f16 workspace (weight shadows + qkv) ----
    _Float16* h16   = (_Float16*)(sup + 128);
    _Float16* qkvh  = h16;                          // 4096*1536
    _Float16* WqkvH = qkvh  + (size_t)N*3*M;        // 2*1536*512
    _Float16* WoH   = WqkvH + (size_t)2*3*M*M;      // 2*512*512
    _Float16* W1H   = WoH   + (size_t)2*M*M;        // 2*2048*512
    _Float16* W2H   = W1H   + (size_t)2*FF*M;       // 2*512*2048
    _Float16* ga1VH = W2H   + (size_t)2*M*FF;       // 256*512
    _Float16* ga1UH = ga1VH + (size_t)Lg*M;         // 256*512

    float* outF  = (float*)d_out;
    float* yprob = outF;            // 16
    float* yhat  = outF + 16;       // 16
    float* a1    = outF + 32;       // 4096
    float* a2    = outF + 4128;     // 128
    float* out2  = outF + 4256;     // 16*512

    // weight f16 shadows (deterministic, every call)
    auto cvt = [&](const float* s, _Float16* d, int n) {
        f32_to_f16<<<(n + 255) / 256, 256, 0, stream>>>(s, d, n);
    };
    cvt(Wqkv, WqkvH, 2*3*M*M);
    cvt(Wo,   WoH,   2*M*M);
    cvt(W1,   W1H,   2*FF*M);
    cvt(W2,   W2H,   2*M*FF);
    cvt(ga1V, ga1VH, Lg*M);
    cvt(ga1U, ga1UH, Lg*M);

    hipMemcpyAsync(X, datas, (size_t)N*M*sizeof(float),
                   hipMemcpyDeviceToDevice, stream);

    for (int l = 0; l < 2; ++l) {
        const _Float16* WqH = WqkvH + (size_t)l*3*M*M;
        const float*    bq  = bqkv  + (size_t)l*3*M;
        const _Float16* Wol = WoH   + (size_t)l*M*M;
        const float*    bol = bo    + (size_t)l*M;
        const _Float16* W1l = W1H   + (size_t)l*FF*M;
        const float*    b1l = b1    + (size_t)l*FF;
        const _Float16* W2l = W2H   + (size_t)l*M*FF;
        const float*    b2l = b2    + (size_t)l*M;

        // qkv (f16 only; attention consumes f16 directly)
        gemm_wmma<<<gemm_blocks(N, 3*M), 256, 0, stream>>>(
            X, WqH, bq, nullptr, nullptr, qkvh, N, 3*M, M, 0);
        attn_flash<<<8 * (N/16), 32, 0, stream>>>(qkvh, AO);
        gemm_wmma<<<gemm_blocks(N, M), 256, 0, stream>>>(
            AO, Wol, bol, X, Y, nullptr, N, M, M, 0);          // proj + residual
        lnorm<<<N, 256, 0, stream>>>(Y, ln1g + (size_t)l*M, ln1b + (size_t)l*M, X);
        gemm_wmma<<<gemm_blocks(N, FF), 256, 0, stream>>>(
            X, W1l, b1l, nullptr, T1, nullptr, N, FF, M, 1);   // relu fused
        gemm_wmma<<<gemm_blocks(N, M), 256, 0, stream>>>(
            T1, W2l, b2l, X, Y, nullptr, N, M, FF, 0);         // + residual
        lnorm<<<N, 256, 0, stream>>>(Y, ln2g + (size_t)l*M, ln2b + (size_t)l*M, X);
    }

    // gated attention level 1
    float* tv = T1;
    float* tu = T1 + (size_t)N*Lg;
    gemm_wmma<<<gemm_blocks(N, Lg), 256, 0, stream>>>(
        X, ga1VH, nullptr, nullptr, tv, nullptr, N, Lg, M, 0);
    gemm_wmma<<<gemm_blocks(N, Lg), 256, 0, stream>>>(
        X, ga1UH, nullptr, nullptr, tu, nullptr, N, Lg, M, 0);
    gated1<<<N, 256, 0, stream>>>(tv, tu, ga1w, w1);
    seg_soft_inner<<<128, 256, 0, stream>>>(w1, Seq_ids, ids, a1, sup);
    pool_inner<<<128, 256, 0, stream>>>(X, a1, Seq_ids, OUTP);

    // gated attention level 2
    gated2<<<128, 256, 0, stream>>>(OUTP, ga2V, ga2U, ga2w, w2v);
    seg_soft_outer<<<1, 128, 0, stream>>>(w2v, sup, a2);
    pool_outer<<<16, 256, 0, stream>>>(OUTP, a2, sup, out2);

    // CNN classifier head
    cnn_head<<<16, 256, 0, stream>>>(out2, conv_w, conv_b, fc_w, fc_b, yprob, yhat);
}
